// BigramLanguageModel_14740327759978
// MI455X (gfx1250) — matrix operand order repaired and checked by reading, not compiled, not run on hardware
//
#include <hip/hip_runtime.h>
#include <hip/hip_bf16.h>
#include <math.h>

// ---------------- model dims ----------------
#define BB 4
#define TT 1024
#define EE 1024
#define HH 16
#define LL 4
#define VV 50257
#define HSZ 64            // head size
#define NTOK (BB * TT)    // 4096 rows

typedef __bf16 v16bf __attribute__((ext_vector_type(16)));
typedef __bf16 v8bf  __attribute__((ext_vector_type(8)));
typedef float  v8f   __attribute__((ext_vector_type(8)));
typedef unsigned int v4u __attribute__((ext_vector_type(4)));
typedef int    v4i   __attribute__((ext_vector_type(4)));
typedef int    v8i   __attribute__((ext_vector_type(8)));

// ---------------- WMMA wrapper ----------------
__device__ __forceinline__ v8f wmma_bf16(v16bf a, v16bf b, v8f c) {
  return __builtin_amdgcn_wmma_f32_16x16x32_bf16(
      false, a, false, b, (short)0, c, false, false);
}

__device__ __forceinline__ v16bf cat8(v8bf lo, v8bf hi) {
  return __builtin_shufflevector(lo, hi, 0, 1, 2, 3, 4, 5, 6, 7,
                                 8, 9, 10, 11, 12, 13, 14, 15);
}

// ---------------- fragment loaders ----------------
// A-fragment (16x32) from row-major f32 (global or LDS), contiguous k:
//   lane holds row; element j <-> k = (j&7) + ((j>>3)<<4) + 8*hi
__device__ __forceinline__ v16bf frag_a_rowmajor(const float* rowp,
                                                 int kb, int hi, float scale) {
  const float4* p0 = reinterpret_cast<const float4*>(rowp + kb + hi * 8);
  const float4* p1 = reinterpret_cast<const float4*>(rowp + kb + 16 + hi * 8);
  float4 f0 = p0[0], f1 = p0[1], f2 = p1[0], f3 = p1[1];
  v16bf r;
  r[0]  = (__bf16)(f0.x * scale); r[1]  = (__bf16)(f0.y * scale);
  r[2]  = (__bf16)(f0.z * scale); r[3]  = (__bf16)(f0.w * scale);
  r[4]  = (__bf16)(f1.x * scale); r[5]  = (__bf16)(f1.y * scale);
  r[6]  = (__bf16)(f1.z * scale); r[7]  = (__bf16)(f1.w * scale);
  r[8]  = (__bf16)(f2.x * scale); r[9]  = (__bf16)(f2.y * scale);
  r[10] = (__bf16)(f2.z * scale); r[11] = (__bf16)(f2.w * scale);
  r[12] = (__bf16)(f3.x * scale); r[13] = (__bf16)(f3.y * scale);
  r[14] = (__bf16)(f3.z * scale); r[15] = (__bf16)(f3.w * scale);
  return r;
}

// B-fragment (32x16), lane's column data: k = (j&7) + ((j>>3)<<4) + 8*hi
__device__ __forceinline__ v16bf frag_b_contig(const float* __restrict__ colp,
                                               int kb, int hi) {
  const float4* p = reinterpret_cast<const float4*>(colp + kb + 8 * hi);
  const float4* q = reinterpret_cast<const float4*>(colp + kb + 16 + 8 * hi);
  float4 f0 = p[0], f1 = p[1], f2 = q[0], f3 = q[1];
  v16bf r;
  r[0]  = (__bf16)f0.x; r[1]  = (__bf16)f0.y; r[2]  = (__bf16)f0.z; r[3]  = (__bf16)f0.w;
  r[4]  = (__bf16)f1.x; r[5]  = (__bf16)f1.y; r[6]  = (__bf16)f1.z; r[7]  = (__bf16)f1.w;
  r[8]  = (__bf16)f2.x; r[9]  = (__bf16)f2.y; r[10] = (__bf16)f2.z; r[11] = (__bf16)f2.w;
  r[12] = (__bf16)f3.x; r[13] = (__bf16)f3.y; r[14] = (__bf16)f3.z; r[15] = (__bf16)f3.w;
  return r;
}

// B-fragment with k strided (unpredicated; V in attention)
__device__ __forceinline__ v16bf frag_b_strided(const float* __restrict__ colp,
                                                long stride, int hi) {
  v16bf r;
#pragma unroll
  for (int j = 0; j < 16; ++j) r[j] = (__bf16)colp[(long)(j + 16 * hi) * stride];
  return r;
}

// ---------------- TDM (Tensor Data Mover) helpers ----------------
__device__ __forceinline__ void tdm_load_2d(const float* gaddr, unsigned lds_off,
                                            unsigned rem0, unsigned rem1,
                                            unsigned tX, unsigned tY,
                                            unsigned long s0) {
  unsigned long long ga = (unsigned long long)gaddr;
  v4u g0;
  g0[0] = 1u;                                   // count=1, user descriptor
  g0[1] = lds_off;                              // lds_addr (bytes)
  g0[2] = (unsigned)(ga & 0xffffffffu);         // global_addr[31:0]
  g0[3] = (unsigned)((ga >> 32) & 0x1ffffffu)   // global_addr[56:32]
        | (2u << 30);                           // type = 2 ("image")
  v8i g1;
  g1[0] = (int)(2u << 16);                      // workgroup_mask=0, data_size=2 (4B)
  g1[1] = (int)((rem0 & 0xffffu) << 16);        // tensor_dim0[15:0]
  g1[2] = (int)((rem0 >> 16) | ((rem1 & 0xffffu) << 16));
  g1[3] = (int)((rem1 >> 16) | (tX << 16));     // dim1 hi | tile_dim0
  g1[4] = (int)(tY & 0xffffu);                  // tile_dim1 (tile_dim2 = 0)
  g1[5] = (int)(unsigned)(s0 & 0xffffffffu);    // tensor_dim0_stride[31:0]
  g1[6] = (int)(unsigned)((s0 >> 32) & 0xffffu);
  g1[7] = 0;
  v4i z4 = {0, 0, 0, 0};
#if defined(__clang_major__) && (__clang_major__ >= 23)
  v8i z8 = {0, 0, 0, 0, 0, 0, 0, 0};
  __builtin_amdgcn_tensor_load_to_lds(g0, g1, z4, z4, z8, 0);
#else
  __builtin_amdgcn_tensor_load_to_lds(g0, g1, z4, z4, 0);
#endif
}

// ---------------- small helpers ----------------
__device__ __forceinline__ float half_group_max(float v) {
#pragma unroll
  for (int off = 8; off >= 1; off >>= 1) v = fmaxf(v, __shfl_xor(v, off, 16));
  return v;
}
__device__ __forceinline__ float half_group_sum(float v) {
#pragma unroll
  for (int off = 8; off >= 1; off >>= 1) v += __shfl_xor(v, off, 16);
  return v;
}
__device__ __forceinline__ float gelu_f(float x) {
  const float c = 0.7978845608028654f;
  return 0.5f * x * (1.0f + tanhf(c * (x + 0.044715f * x * x * x)));
}

// ---------------- kernels ----------------

__global__ __launch_bounds__(256) void embed_kernel(const int* __restrict__ idx,
                                                    const float* __restrict__ tok,
                                                    const float* __restrict__ pos,
                                                    float* __restrict__ x) {
  long i = (long)blockIdx.x * 256 + threadIdx.x;
  if (i >= (long)NTOK * EE) return;
  long bt = i / EE;
  int  e  = (int)(i % EE);
  int  t  = (int)(bt % TT);
  int  tk = idx[bt];
  x[i] = tok[(long)tk * EE + e] + pos[(long)t * EE + e];
}

__global__ __launch_bounds__(256) void layernorm_kernel(const float* __restrict__ x,
                                                        const float* __restrict__ scale,
                                                        const float* __restrict__ bias,
                                                        float* __restrict__ out) {
  __shared__ float sh[2][8];
  int row = blockIdx.x;
  const float* xr = x + (long)row * EE;
  float s = 0.f, ss = 0.f;
  for (int e = threadIdx.x; e < EE; e += 256) {
    float v = xr[e];
    s += v; ss += v * v;
  }
#pragma unroll
  for (int off = 16; off >= 1; off >>= 1) {
    s  += __shfl_xor(s,  off, 32);
    ss += __shfl_xor(ss, off, 32);
  }
  int lane = threadIdx.x & 31, w = threadIdx.x >> 5;
  if (lane == 0) { sh[0][w] = s; sh[1][w] = ss; }
  __syncthreads();
  float ts = 0.f, tss = 0.f;
#pragma unroll
  for (int i = 0; i < 8; ++i) { ts += sh[0][i]; tss += sh[1][i]; }
  float mu  = ts * (1.0f / EE);
  float var = tss * (1.0f / EE) - mu * mu;
  float rs  = rsqrtf(var + 1e-6f);
  for (int e = threadIdx.x; e < EE; e += 256) {
    out[(long)row * EE + e] = (xr[e] - mu) * rs * scale[e] + bias[e];
  }
}

// C[M,N] = epilogue( A[M,K] * B[K,N] ), bf16 WMMA, fp32 accumulate.
// Block tile 64(M) x 128(N), K slab 32, double-buffered, final slab peeled.
// A slab staged by the Tensor Data Mover (f32); B slab via VGPRs into
// transposed bf16 LDS. Each wave computes a 2x2 grid of 16x16 tiles.
// FAST=true: whole 128-col block in range, vector loads, no guards anywhere.
// FAST=false: single ragged edge block (guarded scalar path).
#define LDSTRIDE 40   // bf16 elements per LDS row (16B-aligned, padded)
struct BTile { float b[16]; };

template <bool FAST>
__global__ __launch_bounds__(256) void gemm_bf16_kernel(const float* __restrict__ A,
                                                        const float* __restrict__ Bm,
                                                        const float* __restrict__ bias,
                                                        const float* __restrict__ resid,
                                                        float* __restrict__ C,
                                                        int M, int N, int K, int gelu,
                                                        int colOff) {
  __shared__ float  sAf[2][64][32];          // [buf][m][k]  f32, TDM target (dense)
  __shared__ __bf16 sBt[2][128][LDSTRIDE];   // [buf][n][k]  bf16, transposed
  int t    = threadIdx.x;
  int lane = t & 31, w = t >> 5;
  int hi   = lane >> 4, lr = lane & 15;
  int rowBase = blockIdx.y * 64;
  int colBase = blockIdx.x * 128 + colOff;
  int wrow = (w & 1) * 32;       // wave row group: 0 | 32
  int wcol = (w >> 1) * 32;      // wave col group: 0..96

  int sb_k = t >> 3;             // 0..31 (k row of B)
  int tsub = t & 7;              // 0..7

  auto load_btile = [&](int kb) {
    BTile tl;
    const float* brow = Bm + (long)(kb + sb_k) * N;
    if (FAST) {
      const float4* p = reinterpret_cast<const float4*>(brow + colBase + tsub * 16);
#pragma unroll
      for (int q = 0; q < 4; ++q) {
        float4 f = p[q];
        tl.b[q * 4 + 0] = f.x; tl.b[q * 4 + 1] = f.y;
        tl.b[q * 4 + 2] = f.z; tl.b[q * 4 + 3] = f.w;
      }
    } else {
#pragma unroll
      for (int i = 0; i < 16; ++i) {           // lane-coalesced at each i
        int c = colBase + tsub + 8 * i;
        tl.b[i] = (c < N) ? brow[c] : 0.0f;
      }
    }
    return tl;
  };

  auto store_btile = [&](const BTile& tl, int buf) {
    if (FAST) {
#pragma unroll
      for (int i = 0; i < 16; ++i) sBt[buf][tsub * 16 + i][sb_k] = (__bf16)tl.b[i];
    } else {
#pragma unroll
      for (int i = 0; i < 16; ++i) sBt[buf][tsub + 8 * i][sb_k] = (__bf16)tl.b[i];
    }
  };

  auto issue_tdm_a = [&](int kb, int buf) {
    // flat->LDS: hardware truncates to low 32 bits (ISA 10.2 LDS aperture)
    unsigned lds_off = (unsigned)(unsigned long long)(&sAf[buf][0][0]);
    tdm_load_2d(A + (long)rowBase * K + kb, lds_off,
                (unsigned)(K - kb), (unsigned)(M - rowBase),
                32u, 64u, (unsigned long)K);
  };

  v8f acc[2][2] = {};
  int c0 = wcol + lr, c1 = wcol + 16 + lr;

  auto compute = [&](int buf) {
    v16bf aF0 = frag_a_rowmajor(&sAf[buf][wrow + lr][0],      0, hi, 1.0f);
    v16bf aF1 = frag_a_rowmajor(&sAf[buf][wrow + 16 + lr][0], 0, hi, 1.0f);
    v16bf bF0 = cat8(*reinterpret_cast<const v8bf*>(&sBt[buf][c0][8 * hi]),
                     *reinterpret_cast<const v8bf*>(&sBt[buf][c0][8 * hi + 16]));
    v16bf bF1 = cat8(*reinterpret_cast<const v8bf*>(&sBt[buf][c1][8 * hi]),
                     *reinterpret_cast<const v8bf*>(&sBt[buf][c1][8 * hi + 16]));
    acc[0][0] = wmma_bf16(aF0, bF0, acc[0][0]);
    acc[0][1] = wmma_bf16(aF0, bF1, acc[0][1]);
    acc[1][0] = wmma_bf16(aF1, bF0, acc[1][0]);
    acc[1][1] = wmma_bf16(aF1, bF1, acc[1][1]);
  };

  // ---- prologue: stage slab 0 ----
  if (w == 0) issue_tdm_a(0, 0);
  store_btile(load_btile(0), 0);
  int cur = 0;

  // ---- steady state: no conditionals except uniform w==0 ----
  int kb = 0;
  for (; kb + 32 < K; kb += 32) {
    if (w == 0) __builtin_amdgcn_s_wait_tensorcnt(0);  // A slab `cur` landed
    __syncthreads();                                   // publish slab `cur`
    if (w == 0) issue_tdm_a(kb + 32, cur ^ 1);         // async A for next slab
    BTile nt = load_btile(kb + 32);                    // global B for next slab
    compute(cur);
    store_btile(nt, cur ^ 1);
    cur ^= 1;
  }

  // ---- final slab (peeled) ----
  if (w == 0) __builtin_amdgcn_s_wait_tensorcnt(0);
  __syncthreads();
  compute(cur);

  // ---- epilogue ----
#pragma unroll
  for (int ct = 0; ct < 2; ++ct) {
    int n = colBase + wcol + ct * 16 + lr;
    if (FAST || n < N) {
      float bv = bias ? bias[n] : 0.0f;
#pragma unroll
      for (int rt = 0; rt < 2; ++rt) {
#pragma unroll
        for (int r = 0; r < 8; ++r) {
          int mr = rowBase + wrow + rt * 16 + r + 8 * hi;
          float v = acc[rt][ct][r] + bv;
          if (gelu) v = gelu_f(v);
          if (resid) v += resid[(long)mr * N + n];
          C[(long)mr * N + n] = v;
        }
      }
    }
  }
}

// Flash attention: qkv[B,T,3E] -> y[B,T,E].
__global__ __launch_bounds__(256) void attn_kernel(const float* __restrict__ qkv,
                                                   float* __restrict__ y) {
  __shared__ __bf16 sP[8][16][32];   // per-wave P tile (16 q rows x 32 keys)
  int lane = threadIdx.x & 31;
  int w    = threadIdx.x >> 5;
  int hi   = lane >> 4;
  int lr   = lane & 15;
  int bh = blockIdx.y;
  int b  = bh >> 4;                  // H == 16
  int h  = bh & 15;
  int qbase = blockIdx.x * 128 + w * 16;
  const int S3E = 3 * EE;
  const float* qp = qkv + ((long)b * TT) * S3E + h * HSZ;
  const float* kp = qp + EE;
  const float* vp = qp + 2 * EE;
  const float qscale = 0.125f;       // 1/sqrt(64), folded into Q

  int mq = qbase + lr;
  const float* qrow = qp + (long)mq * S3E;
  v16bf qf0 = frag_a_rowmajor(qrow, 0,  hi, qscale);
  v16bf qf1 = frag_a_rowmajor(qrow, 32, hi, qscale);

  v8f o0 = {}, o1 = {}, o2 = {}, o3 = {};
  float rmax[8], rsum[8];
#pragma unroll
  for (int r = 0; r < 8; ++r) { rmax[r] = -3.0e38f; rsum[r] = 0.0f; }

  for (int jb = 0; jb < qbase + 16; jb += 32) {
    v8f s0 = {}, s1 = {};
    const float* krow0 = kp + (long)(jb + lr) * S3E;
    const float* krow1 = kp + (long)(jb + 16 + lr) * S3E;
    s0 = wmma_bf16(qf0, frag_b_contig(krow0, 0,  hi), s0);
    s0 = wmma_bf16(qf1, frag_b_contig(krow0, 32, hi), s0);
    s1 = wmma_bf16(qf0, frag_b_contig(krow1, 0,  hi), s1);
    s1 = wmma_bf16(qf1, frag_b_contig(krow1, 32, hi), s1);

    int key0 = jb + lr;
    int key1 = jb + 16 + lr;
#pragma unroll
    for (int r = 0; r < 8; ++r) {
      int mrow = qbase + r + 8 * hi;
      float a0 = (key0 <= mrow) ? s0[r] : -3.0e38f;
      float a1 = (key1 <= mrow) ? s1[r] : -3.0e38f;
      float bm = half_group_max(fmaxf(a0, a1));
      float nm = fmaxf(rmax[r], bm);
      float alpha = __expf(rmax[r] - nm);
      float p0 = __expf(a0 - nm);
      float p1 = __expf(a1 - nm);
      float bs = half_group_sum(p0 + p1);
      rsum[r] = rsum[r] * alpha + bs;
      rmax[r] = nm;
      o0[r] *= alpha; o1[r] *= alpha; o2[r] *= alpha; o3[r] *= alpha;
      sP[w][r + 8 * hi][lr]      = (__bf16)p0;
      sP[w][r + 8 * hi][16 + lr] = (__bf16)p1;
    }
    asm volatile("s_wait_dscnt 0" ::: "memory");  // same-wave DS RAW

    v16bf pf;
#pragma unroll
    for (int j = 0; j < 16; ++j) {
      int k = j + 16 * hi;
      pf[j] = sP[w][lr][k];
    }

    const float* vbase = vp + (long)jb * S3E + lr;
    o0 = wmma_bf16(pf, frag_b_strided(vbase + 0,  (long)S3E, hi), o0);
    o1 = wmma_bf16(pf, frag_b_strided(vbase + 16, (long)S3E, hi), o1);
    o2 = wmma_bf16(pf, frag_b_strided(vbase + 32, (long)S3E, hi), o2);
    o3 = wmma_bf16(pf, frag_b_strided(vbase + 48, (long)S3E, hi), o3);
  }

#pragma unroll
  for (int r = 0; r < 8; ++r) {
    int mrow = qbase + r + 8 * hi;
    float inv = 1.0f / rsum[r];
    float* yr = y + ((long)(b * TT + mrow)) * EE + h * HSZ;
    yr[lr]      = o0[r] * inv;
    yr[16 + lr] = o1[r] * inv;
    yr[32 + lr] = o2[r] * inv;
    yr[48 + lr] = o3[r] * inv;
  }
}

__global__ void zero_loss_kernel(float* loss) {
  if (threadIdx.x == 0 && blockIdx.x == 0) *loss = 0.0f;
}

__global__ __launch_bounds__(256) void loss_kernel(const float* __restrict__ logits,
                                                   const int* __restrict__ targets,
                                                   float* __restrict__ loss) {
  __shared__ float sh[8];
  int row = blockIdx.x;
  const float* lg = logits + (size_t)row * VV;
  int lane = threadIdx.x & 31, w = threadIdx.x >> 5;

  float mx = -3.0e38f;
  for (int j = threadIdx.x; j < VV; j += 256) mx = fmaxf(mx, lg[j]);
#pragma unroll
  for (int off = 16; off >= 1; off >>= 1) mx = fmaxf(mx, __shfl_xor(mx, off, 32));
  if (lane == 0) sh[w] = mx;
  __syncthreads();
  float gmx = sh[0];
#pragma unroll
  for (int i = 1; i < 8; ++i) gmx = fmaxf(gmx, sh[i]);
  __syncthreads();

  float se = 0.0f;
  for (int j = threadIdx.x; j < VV; j += 256) se += __expf(lg[j] - gmx);
#pragma unroll
  for (int off = 16; off >= 1; off >>= 1) se += __shfl_xor(se, off, 32);
  if (lane == 0) sh[w] = se;
  __syncthreads();
  if (threadIdx.x == 0) {
    float tot = 0.0f;
#pragma unroll
    for (int i = 0; i < 8; ++i) tot += sh[i];
    int tgt = targets[row];
    float lp = lg[tgt] - gmx - __logf(tot);
    atomicAdd(loss, -lp * (1.0f / (float)NTOK));
  }
}

// ---------------- host-side GEMM dispatch (fast body + ragged edge) ----------
static void launch_gemm(const float* A, const float* Bm, const float* bias,
                        const float* resid, float* C, int M, int N, int K,
                        int gelu, hipStream_t stream) {
  int fullX = N / 128;
  if (fullX > 0) {
    gemm_bf16_kernel<true><<<dim3(fullX, M / 64), 256, 0, stream>>>(
        A, Bm, bias, resid, C, M, N, K, gelu, 0);
  }
  if (N % 128) {
    gemm_bf16_kernel<false><<<dim3(1, M / 64), 256, 0, stream>>>(
        A, Bm, bias, resid, C, M, N, K, gelu, fullX * 128);
  }
}

// ---------------- host orchestration ----------------
extern "C" void kernel_launch(void* const* d_in, const int* in_sizes, int n_in,
                              void* d_out, int out_size, void* d_ws, size_t ws_size,
                              hipStream_t stream) {
  const int*   idx     = (const int*)d_in[0];
  const int*   targets = (const int*)d_in[1];
  const float* tok_emb = (const float*)d_in[2];
  const float* pos_emb = (const float*)d_in[3];
  const float* ln1_s   = (const float*)d_in[4];
  const float* ln1_b   = (const float*)d_in[5];
  const float* Wqkv    = (const float*)d_in[6];
  const float* Wproj   = (const float*)d_in[7];
  const float* ln2_s   = (const float*)d_in[8];
  const float* ln2_b   = (const float*)d_in[9];
  const float* W1      = (const float*)d_in[10];
  const float* b1      = (const float*)d_in[11];
  const float* W2      = (const float*)d_in[12];
  const float* b2      = (const float*)d_in[13];
  const float* lnf_s   = (const float*)d_in[14];
  const float* lnf_b   = (const float*)d_in[15];
  const float* Wlm     = (const float*)d_in[16];
  const float* blm     = (const float*)d_in[17];

  float* logits = (float*)d_out;
  float* loss   = logits + (size_t)NTOK * VV;

  char* ws = (char*)d_ws;
  const size_t SZ_ACT = (size_t)NTOK * EE * sizeof(float);       // 16 MB
  float* x    = (float*)(ws);
  float* hbuf = (float*)(ws + SZ_ACT);
  float* qkvb = (float*)(ws + 2 * SZ_ACT);                       // 48 MB
  float* attb = (float*)(ws + 5 * SZ_ACT);                       // 16 MB
  float* hid  = (float*)(ws + 6 * SZ_ACT);                       // 64 MB

  long tot = (long)NTOK * EE;
  embed_kernel<<<dim3((unsigned)((tot + 255) / 256)), 256, 0, stream>>>(idx, tok_emb, pos_emb, x);

  for (int l = 0; l < LL; ++l) {
    layernorm_kernel<<<NTOK, 256, 0, stream>>>(x, ln1_s + (size_t)l * EE, ln1_b + (size_t)l * EE, hbuf);

    launch_gemm(hbuf, Wqkv + (size_t)l * EE * 3 * EE, nullptr, nullptr, qkvb,
                NTOK, 3 * EE, EE, 0, stream);

    attn_kernel<<<dim3(TT / 128, BB * HH), 256, 0, stream>>>(qkvb, attb);

    launch_gemm(attb, Wproj + (size_t)l * EE * EE, nullptr, x, x,
                NTOK, EE, EE, 0, stream);

    layernorm_kernel<<<NTOK, 256, 0, stream>>>(x, ln2_s + (size_t)l * EE, ln2_b + (size_t)l * EE, hbuf);

    launch_gemm(hbuf, W1 + (size_t)l * EE * 4 * EE, b1 + (size_t)l * 4 * EE, nullptr, hid,
                NTOK, 4 * EE, EE, 1, stream);

    launch_gemm(hid, W2 + (size_t)l * 4 * EE * EE, b2 + (size_t)l * EE, x, x,
                NTOK, EE, 4 * EE, 0, stream);
  }

  layernorm_kernel<<<NTOK, 256, 0, stream>>>(x, lnf_s, lnf_b, hbuf);

  launch_gemm(hbuf, Wlm, blm, nullptr, logits, NTOK, VV, EE, 0, stream);

  zero_loss_kernel<<<1, 1, 0, stream>>>(loss);
  loss_kernel<<<NTOK, 256, 0, stream>>>(logits, targets, loss);
}
